// NDC_70841190580827
// MI455X (gfx1250) — matrix operand order repaired and checked by reading, not compiled, not run on hardware
//
#include <hip/hip_runtime.h>

typedef __attribute__((ext_vector_type(16))) _Float16 v16h;
typedef __attribute__((ext_vector_type(8)))  float    v8f;

__device__ __forceinline__ float tanh_fast(float x) {
#if __has_builtin(__builtin_amdgcn_tanhf)
  return __builtin_amdgcn_tanhf(x);   // v_tanh_f32 (CDNA5 trans op)
#else
  float r;
  asm("v_tanh_f32 %0, %1" : "=v"(r) : "v"(x));
  return r;
#endif
}

__device__ __forceinline__ float readlane_f32(float v, int lane) {
  return __int_as_float(__builtin_amdgcn_readlane(__float_as_int(v), lane));
}

// One wave32 runs the entire serial scan (strict IIR recurrence). Per step:
//   b    = P @ [a0, b1_prev, v]        (rows 0/1 only; wave-uniform scalars)
//   h1   = tanh(W1 @ [b0, dp_r] + b1)  (dp_r folded; redundant per lane;
//                                       lanes 16-31 pre-zeroed => K padding free)
//   u    = W2 @ h1 + b2                (one v_wmma_f32_16x16x32_f16, C = b2)
//   dp_b = W3 @ tanh(u) + b3           (8 lane-local FMAs + 2x v_readlane)
//   out  = 0.5*(b0 + dp_b); carry = (dp_b, P row1 with dp_b)
__global__ __launch_bounds__(32, 1)
void NDC_wdf_scan_kernel(const float* __restrict__ vin,
                         const float* __restrict__ W1,
                         const float* __restrict__ b1,
                         const float* __restrict__ W2,
                         const float* __restrict__ b2,
                         const float* __restrict__ W3,
                         const float* __restrict__ b3,
                         const int*   __restrict__ vsr,
                         float* __restrict__ out,
                         int N)
{
  const int lane = threadIdx.x;

  // ---- WDF port impedances / scattering matrix rows 0 and 1 ----
  const float c1_r = 1.0f / (2.0f * 4.7e-9f * 48000.0f);
  const float vr   = (float)vsr[0];
  const float r0   = c1_r * vr / (c1_r + vr);
  const float g0 = 1.0f / r0, g1 = 1.0f / c1_r, g2 = 1.0f / vr;
  const float gs = 2.0f / (g0 + g1 + g2);
  const float P00 = gs*g0 - 1.0f, P01 = gs*g1,        P02 = gs*g2;
  const float P10 = gs*g0,        P11 = gs*g1 - 1.0f, P12 = gs*g2;
  const float dp_r = (r0 - 1700.0f) * (1.0f / 800.0f);

  // ---- layer-1 constants folded with dp_r; zeroed in lanes 16-31 so the
  //      WMMA B operand's K=16..31 half is produced as zeros for free ----
  const float lm = (lane < 16) ? 1.0f : 0.0f;
  float w1av[16], c1v[16];
  #pragma unroll
  for (int j = 0; j < 16; ++j) {
    w1av[j] = W1[2*j] * lm;
    c1v[j]  = fmaf(W1[2*j + 1], dp_r, b1[j]) * lm;
  }

  // ---- A operand: W2 in WMMA 16-bit A layout (16x32, K=16..31 zero) ----
  // lane m (0-15): halves 0..15 = W2[m, 0..15]; lane m+16: zeros (K=16..31)
  const int row = lane & 15;
  const int kb  = (lane < 16) ? 0 : 8;
  v16h A;
  #pragma unroll
  for (int j = 0; j < 16; ++j) A[j] = (lane < 16) ? (_Float16)W2[row * 16 + j] : (_Float16)0.0f;

  // ---- C operand: b2 broadcast across columns in the f32 C/D layout ----
  const int rb = (lane < 16) ? 0 : 8;     // this lane holds u rows rb..rb+7
  v8f Cb;
  #pragma unroll
  for (int r = 0; r < 8; ++r) Cb[r] = b2[rb + r];

  // ---- W3 split per lane-half to match the D layout ----
  float w3l[8];
  #pragma unroll
  for (int r = 0; r < 8; ++r) w3l[r] = W3[rb + r];
  const float b3s = b3[0];

  float a0 = 0.0f, b1s = 0.0f, ostage = 0.0f;
  float vcur = vin[0];

  for (int t = 0; t < N; ++t) {
    const float v = vcur;
    const int tn = (t + 1 < N) ? (t + 1) : (N - 1);
    vcur = vin[tn];                               // prefetch next sample
    const float kcar = fmaf(P11, b1s, P12 * v);   // carry part, off critical path
    const float dp_a = fmaf(P00, a0, fmaf(P01, b1s, P02 * v));

    // layer 1: every lane computes h1 locally (upper lanes produce zeros)
    v16h B;
    #pragma unroll
    for (int j = 0; j < 16; ++j)
      B[j] = (_Float16)tanh_fast(fmaf(w1av[j], dp_a, c1v[j]));

    // u = W2 @ h1 + b2 in a single matrix op (all 16 output columns identical)
    v8f D = __builtin_amdgcn_wmma_f32_16x16x32_f16(false, A, false, B,
                                                   (short)0, Cb, false, false);

    // layer 3: pairwise dot over this lane's 8 rows
    float p0 = w3l[0] * tanh_fast(D[0]) + w3l[1] * tanh_fast(D[1]);
    float p1 = w3l[2] * tanh_fast(D[2]) + w3l[3] * tanh_fast(D[3]);
    float p2 = w3l[4] * tanh_fast(D[4]) + w3l[5] * tanh_fast(D[5]);
    float p3 = w3l[6] * tanh_fast(D[6]) + w3l[7] * tanh_fast(D[7]);
    float s  = (p0 + p1) + (p2 + p3);

    // cross-half combine without LDS: lanes 0-15 hold rows 0-7 partial,
    // lanes 16-31 hold rows 8-15 partial -> two readlanes, uniform result
    const float dp_b = (readlane_f32(s, 0) + readlane_f32(s, 16)) + b3s;

    const float o = 0.5f * (dp_a + dp_b);
    a0  = dp_b;
    b1s = fmaf(P10, dp_b, kcar);

    // stage one sample per lane; coalesced 32-wide store every 32 steps
    ostage = ((t & 31) == lane) ? o : ostage;
    if ((t & 31) == 31) out[(t & ~31) + lane] = ostage;   // uniform branch
  }
  if (N & 31) {                                           // tail (N%32 != 0)
    if (lane < (N & 31)) out[(N & ~31) + lane] = ostage;
  }
}

extern "C" void kernel_launch(void* const* d_in, const int* in_sizes, int n_in,
                              void* d_out, int out_size, void* d_ws, size_t ws_size,
                              hipStream_t stream) {
  (void)n_in; (void)out_size; (void)d_ws; (void)ws_size;
  const float* vin = (const float*)d_in[0];
  const float* W1  = (const float*)d_in[1];
  const float* b1  = (const float*)d_in[2];
  const float* W2  = (const float*)d_in[3];
  const float* b2  = (const float*)d_in[4];
  const float* W3  = (const float*)d_in[5];
  const float* b3  = (const float*)d_in[6];
  const int*   vsr = (const int*)d_in[7];
  float* out = (float*)d_out;
  const int N = in_sizes[0];
  // Strictly serial IIR: one wave32, minimal per-step latency.
  NDC_wdf_scan_kernel<<<dim3(1), dim3(32), 0, stream>>>(
      vin, W1, b1, W2, b2, W3, b3, vsr, out, N);
}
